// MultiheadAttention_12051678233079
// MI455X (gfx1250) — matrix operand order repaired and checked by reading, not compiled, not run on hardware
//
#include <hip/hip_runtime.h>

#define D_MODEL 2048
#define SEQ     2048
#define BATCH   2
#define NH      16
#define HD      128
#define QKV_N   (3*D_MODEL)
#define MROWS   (BATCH*SEQ)

typedef __attribute__((ext_vector_type(16))) __bf16 v16bf;
typedef __attribute__((ext_vector_type(8)))  float  v8f;

// 16-byte int vector matching the async-to-LDS builtin's pointee type
typedef int vec4i __attribute__((vector_size(16)));
typedef __attribute__((address_space(1))) vec4i g_v4i;
typedef __attribute__((address_space(3))) vec4i l_v4i;

union Frag { v16bf v; uint4 u[2]; };

__device__ __forceinline__ __bf16 f2bf(float f) {
  unsigned u = __builtin_bit_cast(unsigned, f);
  unsigned r = (u + 0x7fffu + ((u >> 16) & 1u)) >> 16;
  unsigned short h = (unsigned short)r;
  return __builtin_bit_cast(__bf16, h);
}
__device__ __forceinline__ float bf2f(__bf16 b) {
  unsigned short h = __builtin_bit_cast(unsigned short, b);
  unsigned u = ((unsigned)h) << 16;
  return __builtin_bit_cast(float, u);
}

__device__ __forceinline__ v8f wmma_bf16(const Frag& a, const Frag& b, v8f c) {
  return __builtin_amdgcn_wmma_f32_16x16x32_bf16(false, a.v, false, b.v,
                                                 (short)0, c, false, false);
}

// ---------- async copy global(16B) -> LDS, with fallback ----------
#if __has_builtin(__builtin_amdgcn_global_load_async_to_lds_b128)
#define HAS_ASYNC_LDS 1
#endif

__device__ __forceinline__ void cp16(const __bf16* g, __bf16* l) {
#if defined(HAS_ASYNC_LDS)
  __builtin_amdgcn_global_load_async_to_lds_b128(
      (g_v4i*)(unsigned long long)g,
      (l_v4i*)(unsigned)(unsigned long long)l, 0, 0);
#else
  *(uint4*)l = *(const uint4*)g;
#endif
}

template <int N>
__device__ __forceinline__ void wait_async() {
#if defined(HAS_ASYNC_LDS)
#if __has_builtin(__builtin_amdgcn_s_wait_asynccnt)
  __builtin_amdgcn_s_wait_asynccnt(N);
#else
  asm volatile("s_wait_asynccnt %0" ::"i"(N) : "memory");
#endif
#endif
}

// ---------------- fp32 -> bf16 convert ----------------
__global__ __launch_bounds__(256) void k_cvt(const float* __restrict__ in,
                                             __bf16* __restrict__ out, size_t n) {
  size_t i = (size_t)blockIdx.x * 256 + threadIdx.x;
  size_t stride = (size_t)gridDim.x * 256;
  for (; i < n; i += stride) out[i] = f2bf(in[i]);
}

// ---------------- block-cooperative NT GEMM with LDS staging ----------------
// A: [MROWS,K] bf16 row-major, W: [N,K] bf16 row-major (K contiguous both).
// Block: 256 threads = 8 waves; tile 128(M) x 128(N); K-step 32, double buffered.
// LDS rows padded to 40 elems (80B = 20 banks) -> conflict-free b128 frag reads.
template <bool CLIP, typename OutT, int N>
__device__ __forceinline__ void gemm_body(const __bf16* __restrict__ A,
                                          const __bf16* __restrict__ W,
                                          const float* __restrict__ bias,
                                          OutT* __restrict__ out) {
  const int K = D_MODEL;
  __shared__ __align__(16) __bf16 sA[2][128 * 40];
  __shared__ __align__(16) __bf16 sB[2][128 * 40];
  int tid = threadIdx.x;
  int w = tid >> 5, lane = tid & 31;
  int half = lane >> 4, l16 = lane & 15;
  int wy = w >> 1, wx = w & 1;
  int bm = blockIdx.x & (MROWS / 128 - 1);
  int bn = blockIdx.x / (MROWS / 128);
  int m0 = bm * 128, n0 = bn * 128;

  // per-thread staging coords: 2 x 16B segments of A and of B per K-step
  int row0 = tid >> 2, s4 = tid & 3;             // rows 0..63
  int row1 = row0 + 64;                          // rows 64..127
  const __bf16* ag0 = A + (size_t)(m0 + row0) * K + s4 * 8;
  const __bf16* ag1 = A + (size_t)(m0 + row1) * K + s4 * 8;
  const __bf16* bg0 = W + (size_t)(n0 + row0) * K + s4 * 8;
  const __bf16* bg1 = W + (size_t)(n0 + row1) * K + s4 * 8;
  int loff0 = row0 * 40 + s4 * 8;
  int loff1 = row1 * 40 + s4 * 8;

  auto stage = [&](int kb, int buf) {
    cp16(ag0 + kb, &sA[buf][loff0]);
    cp16(ag1 + kb, &sA[buf][loff1]);
    cp16(bg0 + kb, &sB[buf][loff0]);
    cp16(bg1 + kb, &sB[buf][loff1]);
  };

  v8f c[2][4] = {};
  const int NK = K / 32;
  stage(0, 0);
  for (int ks = 0; ks < NK; ++ks) {
    int cur = ks & 1;
    if (ks + 1 < NK) {
      stage((ks + 1) * 32, cur ^ 1);
      wait_async<4>();
    } else {
      wait_async<0>();
    }
    __syncthreads();
    const __bf16* Ab = sA[cur];
    const __bf16* Bb = sB[cur];
    Frag a0, a1, b0, b1, b2, b3;
    a0.u[0] = *(const uint4*)(Ab + (wy * 32 + l16) * 40 + half * 16);
    a0.u[1] = *(const uint4*)(Ab + (wy * 32 + l16) * 40 + half * 16 + 8);
    a1.u[0] = *(const uint4*)(Ab + (wy * 32 + 16 + l16) * 40 + half * 16);
    a1.u[1] = *(const uint4*)(Ab + (wy * 32 + 16 + l16) * 40 + half * 16 + 8);
    b0.u[0] = *(const uint4*)(Bb + (wx * 64 + l16) * 40 + half * 16);
    b0.u[1] = *(const uint4*)(Bb + (wx * 64 + l16) * 40 + half * 16 + 8);
    b1.u[0] = *(const uint4*)(Bb + (wx * 64 + 16 + l16) * 40 + half * 16);
    b1.u[1] = *(const uint4*)(Bb + (wx * 64 + 16 + l16) * 40 + half * 16 + 8);
    b2.u[0] = *(const uint4*)(Bb + (wx * 64 + 32 + l16) * 40 + half * 16);
    b2.u[1] = *(const uint4*)(Bb + (wx * 64 + 32 + l16) * 40 + half * 16 + 8);
    b3.u[0] = *(const uint4*)(Bb + (wx * 64 + 48 + l16) * 40 + half * 16);
    b3.u[1] = *(const uint4*)(Bb + (wx * 64 + 48 + l16) * 40 + half * 16 + 8);
    c[0][0] = wmma_bf16(a0, b0, c[0][0]);
    c[0][1] = wmma_bf16(a0, b1, c[0][1]);
    c[0][2] = wmma_bf16(a0, b2, c[0][2]);
    c[0][3] = wmma_bf16(a0, b3, c[0][3]);
    c[1][0] = wmma_bf16(a1, b0, c[1][0]);
    c[1][1] = wmma_bf16(a1, b1, c[1][1]);
    c[1][2] = wmma_bf16(a1, b2, c[1][2]);
    c[1][3] = wmma_bf16(a1, b3, c[1][3]);
    __syncthreads();
  }
#pragma unroll
  for (int j = 0; j < 4; j++) {
    int n = n0 + wx * 64 + j * 16 + l16;
    float bv = bias[n];
#pragma unroll
    for (int i = 0; i < 2; i++) {
#pragma unroll
      for (int r = 0; r < 8; r++) {
        int m = m0 + wy * 32 + i * 16 + r + half * 8;
        float v = c[i][j][r] + bv;
        if constexpr (CLIP) v = fminf(6.0f, fmaxf(-6.0f, v));
        if constexpr (sizeof(OutT) == 2)
          out[(size_t)m * N + n] = (OutT)f2bf(v);
        else
          out[(size_t)m * N + n] = (OutT)v;
      }
    }
  }
}

__global__ __launch_bounds__(256) void k_gemm_qkv(
    const __bf16* __restrict__ A, const __bf16* __restrict__ W,
    const float* __restrict__ bias, __bf16* __restrict__ out) {
  gemm_body<true, __bf16, QKV_N>(A, W, bias, out);
}

__global__ __launch_bounds__(256) void k_gemm_out(
    const __bf16* __restrict__ A, const __bf16* __restrict__ W,
    const float* __restrict__ bias, float* __restrict__ out) {
  gemm_body<false, float, D_MODEL>(A, W, bias, out);
}

// ---------------- LayerNorm(q), LayerNorm(k) -> bf16; v -> Vt[b,h,d,s] bf16 ----
__global__ __launch_bounds__(256) void k_lnqk(
    const __bf16* __restrict__ qkv,
    const float* __restrict__ qg, const float* __restrict__ qb,
    const float* __restrict__ kg, const float* __restrict__ kbv,
    __bf16* __restrict__ qo, __bf16* __restrict__ ko, __bf16* __restrict__ vt) {
  int m = blockIdx.x, t = threadIdx.x;
  const __bf16* row = qkv + (size_t)m * QKV_N;
  float qv[8], kv[8];
  float s0 = 0, s1 = 0, s2 = 0, s3 = 0;
#pragma unroll
  for (int i = 0; i < 8; i++) {
    int e = t + i * 256;
    float q = bf2f(row[e]);
    float k = bf2f(row[D_MODEL + e]);
    qv[i] = q; kv[i] = k;
    s0 += q; s1 += q * q; s2 += k; s3 += k * k;
  }
#pragma unroll
  for (int off = 16; off >= 1; off >>= 1) {
    s0 += __shfl_xor(s0, off, 32);
    s1 += __shfl_xor(s1, off, 32);
    s2 += __shfl_xor(s2, off, 32);
    s3 += __shfl_xor(s3, off, 32);
  }
  __shared__ float red[4][8];
  int w = t >> 5;
  if ((t & 31) == 0) { red[0][w] = s0; red[1][w] = s1; red[2][w] = s2; red[3][w] = s3; }
  __syncthreads();
  s0 = s1 = s2 = s3 = 0;
#pragma unroll
  for (int i = 0; i < 8; i++) { s0 += red[0][i]; s1 += red[1][i]; s2 += red[2][i]; s3 += red[3][i]; }
  const float inv = 1.0f / (float)D_MODEL;
  float muq = s0 * inv, vaq = s1 * inv - muq * muq;
  float muk = s2 * inv, vak = s3 * inv - muk * muk;
  float rq = rsqrtf(vaq + 1e-5f);
  float rk = rsqrtf(vak + 1e-5f);
  int bIdx = m >> 11, s = m & (SEQ - 1);
#pragma unroll
  for (int i = 0; i < 8; i++) {
    int e = t + i * 256;
    qo[(size_t)m * D_MODEL + e] = f2bf((qv[i] - muq) * rq * qg[e] + qb[e]);
    ko[(size_t)m * D_MODEL + e] = f2bf((kv[i] - muk) * rk * kg[e] + kbv[e]);
    float vx = bf2f(row[2 * D_MODEL + e]);
    int h = e >> 7, d = e & 127;
    vt[((size_t)((bIdx * NH + h) * HD + d) << 11) | (size_t)s] = f2bf(vx);
  }
}

// ---------------- Flash attention: one wave per (b, h, 16-row q tile) -----
__global__ __launch_bounds__(128) void k_attn(
    const __bf16* __restrict__ Q, const __bf16* __restrict__ Km,
    const __bf16* __restrict__ Vt, const float* __restrict__ bias,
    const unsigned char* __restrict__ pad, __bf16* __restrict__ Ctx) {
  __shared__ __align__(16) __bf16 plds[4][16 * 40];
  const float scale = 0.08838834764831845f;  // 1/sqrt(128)
  int wIn = threadIdx.x >> 5;
  __bf16* P = plds[wIn];
  int lane = threadIdx.x & 31;
  int half = lane >> 4, l16 = lane & 15;
  int wid = blockIdx.x * 4 + wIn;
  int qt = wid & (SEQ / 16 - 1);
  int h  = (wid >> 7) & (NH - 1);
  int b  = wid >> 11;
  int s0 = qt * 16;

  Frag qa[4];
  const __bf16* qp = Q + (size_t)(b * SEQ + s0 + l16) * D_MODEL + h * HD + half * 16;
#pragma unroll
  for (int c4 = 0; c4 < 4; c4++) {
    qa[c4].u[0] = *(const uint4*)(qp + c4 * 32);
    qa[c4].u[1] = *(const uint4*)(qp + c4 * 32 + 8);
  }

  v8f acc[8] = {};
  float mi[8], li[8];
#pragma unroll
  for (int r = 0; r < 8; r++) { mi[r] = -__builtin_inff(); li[r] = 0.f; }

  const __bf16* kp0 = Km + (size_t)(b * SEQ + l16) * D_MODEL + h * HD + half * 16;
  const __bf16* vp0 = Vt + (size_t)((b * NH + h) * HD + l16) * SEQ + half * 16;
  const float* bp = bias + h * SEQ;
  const unsigned char* pp = pad + b * SEQ;

  for (int kb0 = 0; kb0 <= s0; kb0 += 32) {
    // prefetch next chunk's K rows and V columns (global_prefetch_b8)
    __builtin_prefetch(kp0 + (size_t)(kb0 + 32) * D_MODEL, 0, 1);
    __builtin_prefetch(vp0 + kb0 + 32, 0, 1);
    v8f sc[2];
#pragma unroll
    for (int t = 0; t < 2; t++) {
      v8f s = {};
      const __bf16* kp = kp0 + (size_t)(kb0 + t * 16) * D_MODEL;
#pragma unroll
      for (int c4 = 0; c4 < 4; c4++) {
        Frag kf;
        kf.u[0] = *(const uint4*)(kp + c4 * 32);
        kf.u[1] = *(const uint4*)(kp + c4 * 32 + 8);
        s = wmma_bf16(qa[c4], kf, s);
      }
      int key = kb0 + t * 16 + l16;
      float bv = bp[key];
      int pd = pp[key];
#pragma unroll
      for (int r = 0; r < 8; r++) {
        int qrow = s0 + r + half * 8;
        float v = s[r] * scale + bv;
        s[r] = (pd && key <= qrow) ? v : -__builtin_inff();
      }
      sc[t] = s;
    }
    float mc[8];
#pragma unroll
    for (int r = 0; r < 8; r++) mc[r] = fmaxf(sc[0][r], sc[1][r]);
#pragma unroll
    for (int off = 8; off >= 1; off >>= 1)
#pragma unroll
      for (int r = 0; r < 8; r++) mc[r] = fmaxf(mc[r], __shfl_xor(mc[r], off, 32));
    float al[8], rs[8];
#pragma unroll
    for (int r = 0; r < 8; r++) {
      float mn = fmaxf(mi[r], mc[r]);
      al[r] = __expf(mi[r] - mn);
      mi[r] = mn;
      rs[r] = 0.f;
    }
#pragma unroll
    for (int t = 0; t < 2; t++)
#pragma unroll
      for (int r = 0; r < 8; r++) {
        float p = __expf(sc[t][r] - mi[r]);
        sc[t][r] = p;
        rs[r] += p;
      }
#pragma unroll
    for (int off = 8; off >= 1; off >>= 1)
#pragma unroll
      for (int r = 0; r < 8; r++) rs[r] += __shfl_xor(rs[r], off, 32);
#pragma unroll
    for (int r = 0; r < 8; r++) li[r] = li[r] * al[r] + rs[r];
#pragma unroll
    for (int dt = 0; dt < 8; dt++)
#pragma unroll
      for (int r = 0; r < 8; r++) acc[dt][r] *= al[r];

    // C-layout -> A-layout reshape of P through wave-private LDS
#pragma unroll
    for (int t = 0; t < 2; t++)
#pragma unroll
      for (int r = 0; r < 8; r++)
        P[(r + half * 8) * 40 + t * 16 + l16] = f2bf(sc[t][r]);
    asm volatile("s_wait_dscnt 0" ::: "memory");
    Frag pa;
    pa.u[0] = *(const uint4*)(P + l16 * 40 + half * 16);
    pa.u[1] = *(const uint4*)(P + l16 * 40 + half * 16 + 8);
#pragma unroll
    for (int dt = 0; dt < 8; dt++) {
      const __bf16* vp = vp0 + (size_t)(dt * 16) * SEQ + kb0;
      Frag vf;
      vf.u[0] = *(const uint4*)(vp);
      vf.u[1] = *(const uint4*)(vp + 8);
      acc[dt] = wmma_bf16(pa, vf, acc[dt]);
    }
  }
#pragma unroll
  for (int r = 0; r < 8; r++) li[r] = 1.0f / li[r];
#pragma unroll
  for (int dt = 0; dt < 8; dt++)
#pragma unroll
    for (int r = 0; r < 8; r++) {
      size_t row = (size_t)(b * SEQ + s0 + r + half * 8);
      int col = h * HD + dt * 16 + l16;
      Ctx[row * D_MODEL + col] = f2bf(acc[dt][r] * li[r]);
    }
}

extern "C" void kernel_launch(void* const* d_in, const int* in_sizes, int n_in,
                              void* d_out, int out_size, void* d_ws, size_t ws_size,
                              hipStream_t stream) {
  const float* x         = (const float*)d_in[0];
  const float* attn_bias = (const float*)d_in[1];
  const unsigned char* pad = (const unsigned char*)d_in[2];
  const float* Wqkv_w    = (const float*)d_in[3];
  const float* Wqkv_b    = (const float*)d_in[4];
  const float* q_ln_g    = (const float*)d_in[5];
  const float* q_ln_b    = (const float*)d_in[6];
  const float* k_ln_g    = (const float*)d_in[7];
  const float* k_ln_b    = (const float*)d_in[8];
  const float* out_w     = (const float*)d_in[9];
  const float* out_b     = (const float*)d_in[10];
  float* out = (float*)d_out;

  char* ws = (char*)d_ws;
  size_t off = 0;
  auto take = [&](size_t bytes) {
    char* p = ws + off;
    off += (bytes + 255) & ~(size_t)255;
    return p;
  };
  __bf16* xb   = (__bf16*)take((size_t)MROWS * D_MODEL * 2);
  __bf16* wqb  = (__bf16*)take((size_t)QKV_N * D_MODEL * 2);
  __bf16* owb  = (__bf16*)take((size_t)D_MODEL * D_MODEL * 2);
  __bf16* qkvb = (__bf16*)take((size_t)MROWS * QKV_N * 2);
  __bf16* qbuf = (__bf16*)take((size_t)MROWS * D_MODEL * 2);
  __bf16* kbuf = (__bf16*)take((size_t)MROWS * D_MODEL * 2);
  __bf16* vt   = (__bf16*)take((size_t)MROWS * D_MODEL * 2);
  __bf16* ctxb = (__bf16*)take((size_t)MROWS * D_MODEL * 2);

  k_cvt<<<2048, 256, 0, stream>>>(x, xb, (size_t)MROWS * D_MODEL);
  k_cvt<<<2048, 256, 0, stream>>>(Wqkv_w, wqb, (size_t)QKV_N * D_MODEL);
  k_cvt<<<1024, 256, 0, stream>>>(out_w, owb, (size_t)D_MODEL * D_MODEL);
  k_gemm_qkv<<<(MROWS / 128) * (QKV_N / 128), 256, 0, stream>>>(xb, wqb, Wqkv_b, qkvb);
  k_lnqk<<<MROWS, 256, 0, stream>>>(qkvb, q_ln_g, q_ln_b, k_ln_g, k_ln_b, qbuf, kbuf, vt);
  k_attn<<<(BATCH * NH * (SEQ / 16)) / 4, 128, 0, stream>>>(qbuf, kbuf, vt, attn_bias, pad, ctxb);
  k_gemm_out<<<(MROWS / 128) * (D_MODEL / 128), 256, 0, stream>>>(ctxb, owb, out_b, out);
}
